// MultiheadSelfAttentionEnc_52785148067956
// MI455X (gfx1250) — matrix operand order repaired and checked by reading, not compiled, not run on hardware
//
#include <hip/hip_runtime.h>
#include <hip/hip_bf16.h>

typedef __bf16 bf16;
typedef __attribute__((ext_vector_type(16))) __bf16 v16bf;
typedef __attribute__((ext_vector_type(8)))  __bf16 v8bf;
typedef __attribute__((ext_vector_type(8)))  float  v8f;
typedef __attribute__((ext_vector_type(4))) unsigned int u32x4;
typedef __attribute__((ext_vector_type(8))) int i32x8;
typedef __attribute__((ext_vector_type(4))) int i32x4;

// --------------------------------------------------------------------------
// Tensor Data Mover support detection
// --------------------------------------------------------------------------
#if defined(__has_builtin)
#if __has_builtin(__builtin_amdgcn_tensor_load_to_lds)
#define HAVE_TDM 1
#endif
#endif
#if defined(__has_include)
#if __has_include(<hip/amd_detail/amd_gfx1250_TDM.h>)
#define TDM_SIX_ARGS 1   // therock-10.0 headers => 6-arg builtin
#endif
#endif

__device__ inline uint32_t lds_addr_of(const void* p) {
    return (uint32_t)(uintptr_t)(__attribute__((address_space(3))) const void*)p;
}

#if defined(HAVE_TDM)
// Issue a 2D tile load: LDS[tile_d1][tile_d0] <- global tensor (row stride
// stride0 elements, element size 2 bytes). D# layout per CDNA5 ISA §8.3/8.4.
__device__ inline void tdm_load_2d(uint32_t lds_addr, const void* gptr,
                                   uint32_t tensor_d0, uint32_t tensor_d1,
                                   uint32_t tile_d0, uint32_t tile_d1,
                                   uint64_t stride0) {
    uint64_t ga = (uint64_t)(uintptr_t)gptr;
    u32x4 g0;
    g0[0] = 1u;                                  // count=1 (valid user D#)
    g0[1] = lds_addr;                            // bits 63:32 lds_addr
    g0[2] = (uint32_t)ga;                        // global_addr lo
    g0[3] = (uint32_t)(ga >> 32) | (2u << 30);   // global_addr hi | type=2
    union { uint64_t q[4]; i32x8 v; } g1;
    // q0: workgroup_mask=0 | data_size=1(2B)<<16 | tensor_dim0[15:0]<<48
    g1.q[0] = 0x10000ull | ((uint64_t)(tensor_d0 & 0xFFFFu) << 48);
    // q1: tensor_dim0[31:16] | tensor_dim1<<16 | tile_dim0<<48
    g1.q[1] = (uint64_t)(tensor_d0 >> 16)
            | ((uint64_t)tensor_d1 << 16)
            | ((uint64_t)(tile_d0 & 0xFFFFu) << 48);
    // q2: tile_dim1 | tile_dim2(=0)<<16 | stride0[31:0]<<32
    g1.q[2] = (uint64_t)(tile_d1 & 0xFFFFu)
            | ((stride0 & 0xFFFFFFFFull) << 32);
    // q3: stride0[47:32] | tensor_dim1_stride(=0)<<16
    g1.q[3] = (stride0 >> 32) & 0xFFFFull;
    i32x4 z4 = {0, 0, 0, 0};
#if defined(TDM_SIX_ARGS)
    i32x8 z8 = {0, 0, 0, 0, 0, 0, 0, 0};
    __builtin_amdgcn_tensor_load_to_lds(g0, g1.v, z4, z4, z8, 0);
#else
    __builtin_amdgcn_tensor_load_to_lds(g0, g1.v, z4, z4, 0);
#endif
}
#endif  // HAVE_TDM

// ---------------------------------------------------------------------------
// WMMA fragment helpers (CDNA5 16x16x32 bf16, wave32)
// A layout: lane L (m = L&15, half = L>>4): elements 0..15 = K 16*half+0..15
//                                            (contiguous)
// B layout (stored transposed as BT[n][k]): lane L (n = L&15, koff = 16*(L>>4)):
//           elements 0..15 = K koff..koff+15 (contiguous)
// C/D layout: lane L (n = L&15): element r = (M = r + 8*(L>>4), N = n)
// ---------------------------------------------------------------------------
__device__ inline v16bf load_a_frag(const bf16* base, int ld) {
    int lane = threadIdx.x & 31;
    int hi   = lane >> 4;
    int m    = lane & 15;
    const bf16* p = base + m * ld + hi * 16;
    v8bf lo = *(const v8bf*)(p);
    v8bf hv = *(const v8bf*)(p + 8);
    v16bf r;
#pragma unroll
    for (int e = 0; e < 8; ++e) { r[e] = lo[e]; r[8 + e] = hv[e]; }
    return r;
}

__device__ inline v16bf load_bt_frag(const bf16* base, int ld) {
    int lane = threadIdx.x & 31;
    int n    = lane & 15;
    int koff = (lane >> 4) * 16;
    return *(const v16bf*)(base + n * ld + koff);
}

__device__ inline v8f wmma_bf16(v16bf a, v16bf b, v8f c) {
    return __builtin_amdgcn_wmma_f32_16x16x32_bf16(
        false, a, false, b, (short)0, c, false, false);
}

// ---------------------------------------------------------------------------
// fp32 -> bf16 conversion (vector-4, grid-stride)
// ---------------------------------------------------------------------------
__global__ void f32_to_bf16_kernel(const float* __restrict__ in,
                                   bf16* __restrict__ out, size_t n4) {
    size_t i = (size_t)blockIdx.x * blockDim.x + threadIdx.x;
    size_t stride = (size_t)gridDim.x * blockDim.x;
    for (; i < n4; i += stride) {
        float4 f = *(const float4*)(in + 4 * i);
        bf16* o = out + 4 * i;
        o[0] = (bf16)f.x; o[1] = (bf16)f.y; o[2] = (bf16)f.z; o[3] = (bf16)f.w;
    }
}

// ---------------------------------------------------------------------------
// Y[m,o] = sum_k A[m,k] * W[o,k] + bias[o]   (torch Linear: x @ W.T + b)
// Block 256 threads / 8 waves; tile 128(M) x 64(N); K-chunk 64,
// double-buffered through LDS via TDM (fallback: cooperative copy).
// Fragments preloaded before the WMMA burst so ds_loads clause together.
// ---------------------------------------------------------------------------
#define GEMM_BM 128
#define GEMM_BN 64
#define GEMM_BK 64

__global__ __launch_bounds__(256) void gemm_bf16_staged(
    const bf16* __restrict__ A, const bf16* __restrict__ W,
    const float* __restrict__ bias, float* __restrict__ Y,
    int M, int N, int K) {
    __shared__ __align__(128) bf16 sA[2][GEMM_BM * GEMM_BK];  // [m][k]
    __shared__ __align__(128) bf16 sB[2][GEMM_BN * GEMM_BK];  // [n][k]

    int wave  = threadIdx.x >> 5;
    int lane  = threadIdx.x & 31;
    int m0blk = blockIdx.y * GEMM_BM;
    int n0blk = blockIdx.x * GEMM_BN;
    int nk    = K / GEMM_BK;

    v8f acc[4];
#pragma unroll
    for (int t = 0; t < 4; ++t)
#pragma unroll
        for (int r = 0; r < 8; ++r) acc[t][r] = 0.f;

#if defined(HAVE_TDM)
    if (threadIdx.x < 32) {
        tdm_load_2d(lds_addr_of(&sA[0][0]), A + (size_t)m0blk * K,
                    (uint32_t)K, (uint32_t)M, GEMM_BK, GEMM_BM, (uint64_t)K);
        tdm_load_2d(lds_addr_of(&sB[0][0]), W + (size_t)n0blk * K,
                    (uint32_t)K, (uint32_t)N, GEMM_BK, GEMM_BN, (uint64_t)K);
    }
#endif

    for (int kc = 0; kc < nk; ++kc) {
        int cur = kc & 1;
        int k0  = kc * GEMM_BK;
#if defined(HAVE_TDM)
        if (threadIdx.x < 32) {
            if (kc + 1 < nk) {
                int nxt = 1 - cur;
                int k1  = k0 + GEMM_BK;
                tdm_load_2d(lds_addr_of(&sA[nxt][0]),
                            A + (size_t)m0blk * K + k1,
                            (uint32_t)K, (uint32_t)M, GEMM_BK, GEMM_BM,
                            (uint64_t)K);
                tdm_load_2d(lds_addr_of(&sB[nxt][0]),
                            W + (size_t)n0blk * K + k1,
                            (uint32_t)K, (uint32_t)N, GEMM_BK, GEMM_BN,
                            (uint64_t)K);
                __builtin_amdgcn_s_wait_tensorcnt(2);  // current buffer done
            } else {
                __builtin_amdgcn_s_wait_tensorcnt(0);
            }
        }
        __syncthreads();
#else
        __syncthreads();
        {   // cooperative stage: 256 threads, 16B vectors
            int t = threadIdx.x;
            int cseg = (t & 7) * 8;
            int rbase = t >> 3;  // 0..31
#pragma unroll
            for (int p = 0; p < 4; ++p) {
                int row = rbase + p * 32;
                *(v8bf*)(&sA[cur][row * GEMM_BK + cseg]) =
                    *(const v8bf*)(A + (size_t)(m0blk + row) * K + k0 + cseg);
            }
#pragma unroll
            for (int p = 0; p < 2; ++p) {
                int row = rbase + p * 32;
                *(v8bf*)(&sB[cur][row * GEMM_BK + cseg]) =
                    *(const v8bf*)(W + (size_t)(n0blk + row) * K + k0 + cseg);
            }
        }
        __syncthreads();
#endif
        // preload all fragments, then WMMA burst (lets ds_loads clause and
        // waits become partial instead of load->wait0->wmma round trips)
        int m0 = wave * 16;
        v16bf afr[2];
        v16bf bfr[2][4];
#pragma unroll
        for (int kk = 0; kk < 2; ++kk) {
            afr[kk] = load_a_frag(&sA[cur][m0 * GEMM_BK + kk * 32], GEMM_BK);
#pragma unroll
            for (int t = 0; t < 4; ++t)
                bfr[kk][t] = load_bt_frag(
                    &sB[cur][(t * 16) * GEMM_BK + kk * 32], GEMM_BK);
        }
#pragma unroll
        for (int kk = 0; kk < 2; ++kk)
#pragma unroll
            for (int t = 0; t < 4; ++t)
                acc[t] = wmma_bf16(afr[kk], bfr[kk][t], acc[t]);
#if defined(HAVE_TDM)
        __syncthreads();  // reads done before next TDM overwrite
#endif
    }

    int hi = lane >> 4, n = lane & 15;
    int mrow = m0blk + wave * 16;
#pragma unroll
    for (int t = 0; t < 4; ++t) {
        int col = n0blk + t * 16 + n;
        float bv = bias[col];
#pragma unroll
        for (int r = 0; r < 8; ++r) {
            int row = mrow + r + 8 * hi;
            Y[(size_t)row * N + col] = acc[t][r] + bv;
        }
    }
}

// ---------------------------------------------------------------------------
// Fused RMSNorm(+elementwise scale) + RoPE + bf16 cast for q/k; bf16 cast for v.
// ---------------------------------------------------------------------------
__global__ __launch_bounds__(256) void rmsnorm_rope_kernel(
    const float* __restrict__ qf, const float* __restrict__ kf,
    const float* __restrict__ vf, const float* __restrict__ rope,
    const float* __restrict__ q_scale, const float* __restrict__ k_scale,
    bf16* __restrict__ qb, bf16* __restrict__ kb, bf16* __restrict__ vb,
    int S, int H) {
    const int C = H * 128;
    int wid = blockIdx.x * 8 + (threadIdx.x >> 5);
    if (wid >= S * H) return;
    int lane = threadIdx.x & 31;
    int s = wid / H, h = wid % H;
    int d0 = lane * 4;
    size_t off = (size_t)s * C + h * 128 + d0;

    float4 q = *(const float4*)(qf + off);
    float4 k = *(const float4*)(kf + off);
    float4 v = *(const float4*)(vf + off);

    float sq = q.x * q.x + q.y * q.y + q.z * q.z + q.w * q.w;
    float sk = k.x * k.x + k.y * k.y + k.z * k.z + k.w * k.w;
#pragma unroll
    for (int m = 16; m >= 1; m >>= 1) {
        sq += __shfl_xor(sq, m, 32);
        sk += __shfl_xor(sk, m, 32);
    }
    const float eps = 1.1920928955078125e-07f;
    float rq = rsqrtf(sq * (1.f / 128.f) + eps);
    float rk = rsqrtf(sk * (1.f / 128.f) + eps);

    float4 qs = *(const float4*)(q_scale + d0);
    float4 ks = *(const float4*)(k_scale + d0);
    float qn0 = q.x * rq * qs.x, qn1 = q.y * rq * qs.y;
    float qn2 = q.z * rq * qs.z, qn3 = q.w * rq * qs.w;
    float kn0 = k.x * rk * ks.x, kn1 = k.y * rk * ks.y;
    float kn2 = k.z * rk * ks.z, kn3 = k.w * rk * ks.w;

    const float4 r0 = *(const float4*)(rope + ((size_t)s * 64 + (d0 >> 1)) * 4);
    const float4 r1 = *(const float4*)(rope + ((size_t)s * 64 + (d0 >> 1) + 1) * 4);

    qb[off + 0] = (bf16)(r0.x * qn0 + r0.y * qn1);
    qb[off + 1] = (bf16)(r0.z * qn0 + r0.w * qn1);
    qb[off + 2] = (bf16)(r1.x * qn2 + r1.y * qn3);
    qb[off + 3] = (bf16)(r1.z * qn2 + r1.w * qn3);

    kb[off + 0] = (bf16)(r0.x * kn0 + r0.y * kn1);
    kb[off + 1] = (bf16)(r0.z * kn0 + r0.w * kn1);
    kb[off + 2] = (bf16)(r1.x * kn2 + r1.y * kn3);
    kb[off + 3] = (bf16)(r1.z * kn2 + r1.w * kn3);

    vb[off + 0] = (bf16)v.x; vb[off + 1] = (bf16)v.y;
    vb[off + 2] = (bf16)v.z; vb[off + 3] = (bf16)v.w;
}

// ---------------------------------------------------------------------------
// Flash attention: grid (S/64, H), block 128 (4 waves). Wave owns 16 q-rows.
// K tile staged row-major via TDM (fallback: cooperative loads); V tile
// manually transposed into LDS (PV B-operand); P transposed through LDS.
// ---------------------------------------------------------------------------
__global__ __launch_bounds__(128) void flash_attn_kernel(
    const bf16* __restrict__ qb, const bf16* __restrict__ kb,
    const bf16* __restrict__ vb, bf16* __restrict__ aout, int S, int H) {
    const int C = H * 128;
    __shared__ __align__(128) bf16 Kt[32 * 128];    // [key][d]
    __shared__ __align__(128) bf16 Vt[128 * 32];    // [d][key]  (transposed)
    __shared__ __align__(32)  bf16 Ps[4][16 * 32];  // per-wave P tile [m][key]

    int wave = threadIdx.x >> 5;
    int lane = threadIdx.x & 31;
    int h    = blockIdx.y;
    int q0   = blockIdx.x * 64 + wave * 16;

    v16bf aq[4];
#pragma unroll
    for (int kk = 0; kk < 4; ++kk)
        aq[kk] = load_a_frag(qb + (size_t)q0 * C + h * 128 + kk * 32, C);

    v8f accO[8];
    float mrow[8], lrow[8];
#pragma unroll
    for (int t = 0; t < 8; ++t)
#pragma unroll
        for (int r = 0; r < 8; ++r) accO[t][r] = 0.f;
#pragma unroll
    for (int r = 0; r < 8; ++r) { mrow[r] = -1e30f; lrow[r] = 0.f; }

    const float scale = 0.088388347648318447f;  // 1/sqrt(128)
    int nkb = S / 32;
    for (int j = 0; j < nkb; ++j) {
        __syncthreads();
#if defined(HAVE_TDM)
        if (threadIdx.x < 32) {   // K tile: 32 keys x 128 d, row stride C
            tdm_load_2d(lds_addr_of(&Kt[0]),
                        kb + (size_t)(j * 32) * C + h * 128,
                        (uint32_t)C, (uint32_t)S, 128, 32, (uint64_t)C);
        }
        {   // V tile transposed, manual (TDM cannot transpose)
            int key  = threadIdx.x >> 2;
            int dseg = (threadIdx.x & 3) * 32;
            const bf16* vr = vb + (size_t)(j * 32 + key) * C + h * 128 + dseg;
#pragma unroll
            for (int c = 0; c < 4; ++c) {
                v8bf vv = *(const v8bf*)(vr + c * 8);
#pragma unroll
                for (int e = 0; e < 8; ++e)
                    Vt[(dseg + c * 8 + e) * 32 + key] = vv[e];
            }
        }
        if (threadIdx.x < 32) __builtin_amdgcn_s_wait_tensorcnt(0);
        __syncthreads();
#else
        {   // cooperative stage of K (row-major) and V (transposed)
            int key  = threadIdx.x >> 2;
            int dseg = (threadIdx.x & 3) * 32;
            const bf16* kr = kb + (size_t)(j * 32 + key) * C + h * 128 + dseg;
            const bf16* vr = vb + (size_t)(j * 32 + key) * C + h * 128 + dseg;
#pragma unroll
            for (int c = 0; c < 4; ++c) {
                v8bf kv = *(const v8bf*)(kr + c * 8);
                *(v8bf*)(&Kt[key * 128 + dseg + c * 8]) = kv;
                v8bf vv = *(const v8bf*)(vr + c * 8);
#pragma unroll
                for (int e = 0; e < 8; ++e)
                    Vt[(dseg + c * 8 + e) * 32 + key] = vv[e];
            }
        }
        __syncthreads();
#endif

        v8f sc[2];
#pragma unroll
        for (int st = 0; st < 2; ++st) {
#pragma unroll
            for (int r = 0; r < 8; ++r) sc[st][r] = 0.f;
#pragma unroll
            for (int kk = 0; kk < 4; ++kk) {
                v16bf bK = load_bt_frag(&Kt[st * 16 * 128 + kk * 32], 128);
                sc[st] = wmma_bf16(aq[kk], bK, sc[st]);
            }
        }

        float alpha[8];
#pragma unroll
        for (int r = 0; r < 8; ++r) {
            float s0 = sc[0][r] * scale, s1 = sc[1][r] * scale;
            float mx = fmaxf(s0, s1);
#pragma unroll
            for (int m = 8; m >= 1; m >>= 1) mx = fmaxf(mx, __shfl_xor(mx, m, 16));
            float mnew = fmaxf(mrow[r], mx);
            float a  = __expf(mrow[r] - mnew);
            float p0 = __expf(s0 - mnew);
            float p1 = __expf(s1 - mnew);
            float ps = p0 + p1;
#pragma unroll
            for (int m = 8; m >= 1; m >>= 1) ps += __shfl_xor(ps, m, 16);
            lrow[r] = lrow[r] * a + ps;
            mrow[r] = mnew;
            alpha[r] = a;
            int M = r + 8 * (lane >> 4);
            Ps[wave][M * 32 + (lane & 15)]      = (bf16)p0;
            Ps[wave][M * 32 + 16 + (lane & 15)] = (bf16)p1;
        }
#pragma unroll
        for (int t = 0; t < 8; ++t)
#pragma unroll
            for (int r = 0; r < 8; ++r) accO[t][r] *= alpha[r];

        v16bf ap = load_a_frag(&Ps[wave][0], 32);
#pragma unroll
        for (int t = 0; t < 8; ++t) {
            v16bf bV = load_bt_frag(&Vt[t * 16 * 32], 32);
            accO[t] = wmma_bf16(ap, bV, accO[t]);
        }
    }

    int hi = lane >> 4, n = lane & 15;
#pragma unroll
    for (int t = 0; t < 8; ++t)
#pragma unroll
        for (int r = 0; r < 8; ++r) {
            int row = q0 + r + 8 * hi;
            float o = accO[t][r] / lrow[r];
            aout[(size_t)row * C + h * 128 + t * 16 + n] = (bf16)o;
        }
}

// ---------------------------------------------------------------------------
// Host-side launch
// ---------------------------------------------------------------------------
extern "C" void kernel_launch(void* const* d_in, const int* in_sizes, int n_in,
                              void* d_out, int out_size, void* d_ws, size_t ws_size,
                              hipStream_t stream) {
    const float* x      = (const float*)d_in[0];
    const float* rope   = (const float*)d_in[1];
    const float* Wq     = (const float*)d_in[2];
    const float* bq     = (const float*)d_in[3];
    const float* Wk     = (const float*)d_in[4];
    const float* bk     = (const float*)d_in[5];
    const float* Wv     = (const float*)d_in[6];
    const float* bv     = (const float*)d_in[7];
    const float* Wo     = (const float*)d_in[8];
    const float* bo     = (const float*)d_in[9];
    const float* qscale = (const float*)d_in[10];
    const float* kscale = (const float*)d_in[11];
    float* out = (float*)d_out;

    const int C = in_sizes[3];
    const int S = in_sizes[0] / C;
    const int H = C / 128;
    const size_t SC = (size_t)S * C;
    const size_t CC = (size_t)C * C;

    char* ws = (char*)d_ws;
    size_t off = 0;
    auto walloc = [&](size_t bytes) -> void* {
        void* p = ws + off;
        off = (off + bytes + 255) & ~(size_t)255;
        return p;
    };
    bf16* xb   = (bf16*)walloc(SC * 2);
    bf16* Wqb  = (bf16*)walloc(CC * 2);
    bf16* Wkb  = (bf16*)walloc(CC * 2);
    bf16* Wvb  = (bf16*)walloc(CC * 2);
    bf16* Wob  = (bf16*)walloc(CC * 2);
    float* qf  = (float*)walloc(SC * 4);
    float* kf  = (float*)walloc(SC * 4);
    float* vf  = (float*)walloc(SC * 4);
    bf16* qb   = (bf16*)walloc(SC * 2);
    bf16* kb   = (bf16*)walloc(SC * 2);
    bf16* vb   = (bf16*)walloc(SC * 2);
    bf16* aout = (bf16*)walloc(SC * 2);

    f32_to_bf16_kernel<<<2048, 256, 0, stream>>>(x,  xb,  SC / 4);
    f32_to_bf16_kernel<<<2048, 256, 0, stream>>>(Wq, Wqb, CC / 4);
    f32_to_bf16_kernel<<<2048, 256, 0, stream>>>(Wk, Wkb, CC / 4);
    f32_to_bf16_kernel<<<2048, 256, 0, stream>>>(Wv, Wvb, CC / 4);
    f32_to_bf16_kernel<<<2048, 256, 0, stream>>>(Wo, Wob, CC / 4);

    dim3 ggrid(C / GEMM_BN, S / GEMM_BM);
    gemm_bf16_staged<<<ggrid, 256, 0, stream>>>(xb, Wqb, bq, qf, S, C, C);
    gemm_bf16_staged<<<ggrid, 256, 0, stream>>>(xb, Wkb, bk, kf, S, C, C);
    gemm_bf16_staged<<<ggrid, 256, 0, stream>>>(xb, Wvb, bv, vf, S, C, C);

    rmsnorm_rope_kernel<<<(S * H) / 8, 256, 0, stream>>>(
        qf, kf, vf, rope, qscale, kscale, qb, kb, vb, S, H);

    dim3 agrid(S / 64, H);
    flash_attn_kernel<<<agrid, 128, 0, stream>>>(qb, kb, vb, aout, S, H);

    gemm_bf16_staged<<<ggrid, 256, 0, stream>>>(aout, Wob, bo, out, S, C, C);
}
